// MultiHeadAttentionEdgeEmb1_34256659153220
// MI455X (gfx1250) — matrix operand order repaired and checked by reading, not compiled, not run on hardware
//
#include <hip/hip_runtime.h>

typedef __attribute__((ext_vector_type(16))) _Float16 v16h;
typedef __attribute__((ext_vector_type(8)))  _Float16 v8h;
typedef __attribute__((ext_vector_type(4)))  _Float16 v4h;
typedef __attribute__((ext_vector_type(8)))  float    v8f;

#define WAVES_PER_BLOCK 4
#define TPB (WAVES_PER_BLOCK * 32)
#define MT 2                               // 16-row M-tiles per wave
#define EDGES_PER_WAVE (MT * 16)           // 32 edges per wave

__device__ __forceinline__ v8f wmma_f16(v16h a, v16h b, v8f c) {
  // (neg_a, A, neg_b, B, c_mod, C, reuse_a, reuse_b)
  return __builtin_amdgcn_wmma_f32_16x16x32_f16(false, a, false, b, (short)0, c, false, false);
}

__device__ __forceinline__ float sigmoid_f(float x) {
  return 1.0f / (1.0f + __expf(-x));
}

// A fragment (16x32 MxK, f16) from a row-major [16][128] f16 LDS window.
// ISA layout: lanes 0-15 row M=lane hold K {0..7,16..23}; lanes 16-31 hold K {8..15,24..31}.
__device__ __forceinline__ v16h load_a_frag(const _Float16* buf, int lane, int kt) {
  const int r  = lane & 15;
  const int lh = lane >> 4;
  const _Float16* p = buf + r * 128 + kt * 32 + lh * 8;
  v8h lo = *(const v8h*)(p);
  v8h hi = *(const v8h*)(p + 16);
  v16h a;
#pragma unroll
  for (int i = 0; i < 8; ++i) { a[i] = lo[i]; a[i + 8] = hi[i]; }
  return a;
}

// B fragment (32x16 KxN): pre-swizzled in LDS; each lane reads 16 contiguous halfs (32 B).
__device__ __forceinline__ v16h load_b_frag(const _Float16* w, int lane, int kt, int nt) {
  return *(const v16h*)(w + ((((kt << 3) + nt) << 5) + lane) * 16);
}

// Convert row-major [K][128] fp32 weight -> per-lane f16 B fragments in LDS.
// B lane layout: N = nt*16 + (lane&15); K = kt*32 + (lane>=16 ? 8 : 0) + (h&7) + (h>=8 ? 16 : 0).
__device__ __forceinline__ void prep_weight(const float* __restrict__ W, _Float16* dst, int K) {
  const int total = K * 128;
  for (int i = threadIdx.x; i < total; i += TPB) {
    const int h  = i & 15;
    const int l  = (i >> 4) & 31;
    const int f  = i >> 9;
    const int nt = f & 7;
    const int kt = f >> 3;
    const int k  = (kt << 5) + ((l >> 4) << 3) + (h & 7) + ((h >> 3) << 4);
    const int n  = (nt << 4) + (l & 15);
    dst[i] = (_Float16)W[k * 128 + n];
  }
}

// In-place gated projection on one wave-private [32][128] row buffer:
// buf <- (f16) ( sigmoid(buf @ W) * buf ).  B fragments reused across both M-tiles.
__device__ __forceinline__ void gate_gemm(_Float16* buf, const _Float16* sW, int lane) {
  const int cl = lane & 15, lh = lane >> 4;
  v16h a[MT][4];
#pragma unroll
  for (int mt = 0; mt < MT; ++mt)
#pragma unroll
    for (int kt = 0; kt < 4; ++kt) a[mt][kt] = load_a_frag(buf + mt * 2048, lane, kt);

#pragma unroll
  for (int nt = 0; nt < 8; ++nt) {
    v8f c0 = {}, c1 = {};
#pragma unroll
    for (int kt = 0; kt < 4; ++kt) {
      v16h b = load_b_frag(sW, lane, kt, nt);
      c0 = wmma_f16(a[0][kt], b, c0);
      c1 = wmma_f16(a[1][kt], b, c1);
    }
    const int n = (nt << 4) + cl;
#pragma unroll
    for (int v = 0; v < 8; ++v) {            // C layout: M = v + lh*8 (+16 for tile 1)
      const int r0 = (v + lh * 8) * 128 + n;
      const int r1 = r0 + 16 * 128;
      float x0 = (float)buf[r0];
      float x1 = (float)buf[r1];
      buf[r0] = (_Float16)(sigmoid_f(c0[v]) * x0);
      buf[r1] = (_Float16)(sigmoid_f(c1[v]) * x1);
    }
  }
}

__global__ __launch_bounds__(TPB)
void MHAEdgeEmb_kernel(const float* __restrict__ emb, const int* __restrict__ eidx,
                       const float* __restrict__ Wsrc, const float* __restrict__ Wtgt,
                       const float* __restrict__ Wproj, const float* __restrict__ bproj,
                       const float* __restrict__ Wmlp, const float* __restrict__ bmlp,
                       float* __restrict__ out, int E, int ntiles) {
  __shared__ __align__(32) _Float16 sW[256 * 128];                                   // 64 KB
  __shared__ __align__(32) _Float16 sRow[WAVES_PER_BLOCK][2][EDGES_PER_WAVE * 128];  // 64 KB
  __shared__ float sBproj[128];
  __shared__ float sBmlp[128];

  const int tid  = threadIdx.x;
  const int lane = tid & 31;
  const int wave = tid >> 5;
  const int cl = lane & 15, lh = lane >> 4;
  const int tile = blockIdx.x * WAVES_PER_BLOCK + wave;
  const bool active = tile < ntiles;
  const int e0 = tile * EDGES_PER_WAVE;
  _Float16* bufS = sRow[wave][0];
  _Float16* bufT = sRow[wave][1];

  for (int i = tid; i < 128; i += TPB) { sBproj[i] = bproj[i]; sBmlp[i] = bmlp[i]; }

  // ---- Phase 1: gather rows (fp32 -> f16 LDS) + src gate GEMM ----
  prep_weight(Wsrc, sW, 128);
  __syncthreads();
  if (active) {
#pragma unroll 4
    for (int m = 0; m < EDGES_PER_WAVE; ++m) {
      int e = e0 + m; if (e >= E) e = E - 1;
      const float4 vs = ((const float4*)(emb + (size_t)eidx[e] * 128))[lane];
      const float4 vt = ((const float4*)(emb + (size_t)eidx[E + e] * 128))[lane];
      v4h hs, ht;
      hs[0] = (_Float16)vs.x; hs[1] = (_Float16)vs.y; hs[2] = (_Float16)vs.z; hs[3] = (_Float16)vs.w;
      ht[0] = (_Float16)vt.x; ht[1] = (_Float16)vt.y; ht[2] = (_Float16)vt.z; ht[3] = (_Float16)vt.w;
      *(v4h*)(bufS + m * 128 + lane * 4) = hs;
      *(v4h*)(bufT + m * 128 + lane * 4) = ht;
    }
    gate_gemm(bufS, sW, lane);
  }
  __syncthreads();

  // ---- Phase 2: tgt gate GEMM ----
  prep_weight(Wtgt, sW, 128);
  __syncthreads();
  if (active) gate_gemm(bufT, sW, lane);
  __syncthreads();

  // ---- Phase 3: [attn_s | attn_t] @ W_proj + b_proj, ReLU -> bufS (in place) ----
  prep_weight(Wproj, sW, 256);
  __syncthreads();
  if (active) {
    v16h a[MT][8];                       // kt 0..3 from attn_s, 4..7 from attn_t
#pragma unroll
    for (int mt = 0; mt < MT; ++mt)
#pragma unroll
      for (int kt = 0; kt < 4; ++kt) {
        a[mt][kt]     = load_a_frag(bufS + mt * 2048, lane, kt);
        a[mt][kt + 4] = load_a_frag(bufT + mt * 2048, lane, kt);
      }
#pragma unroll
    for (int nt = 0; nt < 8; ++nt) {
      v8f c0 = {}, c1 = {};
#pragma unroll
      for (int kt = 0; kt < 8; ++kt) {
        v16h b = load_b_frag(sW, lane, kt, nt);
        c0 = wmma_f16(a[0][kt], b, c0);
        c1 = wmma_f16(a[1][kt], b, c1);
      }
      const int n = (nt << 4) + cl;
      const float bb = sBproj[n];
#pragma unroll
      for (int v = 0; v < 8; ++v) {
        float y0 = c0[v] + bb;
        float y1 = c1[v] + bb;
        bufS[(v + lh * 8) * 128 + n]        = (_Float16)(y0 > 0.0f ? y0 : 0.0f);
        bufS[(16 + v + lh * 8) * 128 + n]   = (_Float16)(y1 > 0.0f ? y1 : 0.0f);
      }
    }
  }
  __syncthreads();

  // ---- Phase 4: relu @ W_mlp + b_mlp -> global out (fp32) ----
  prep_weight(Wmlp, sW, 128);
  __syncthreads();
  if (active) {
    v16h a[MT][4];
#pragma unroll
    for (int mt = 0; mt < MT; ++mt)
#pragma unroll
      for (int kt = 0; kt < 4; ++kt) a[mt][kt] = load_a_frag(bufS + mt * 2048, lane, kt);
#pragma unroll
    for (int nt = 0; nt < 8; ++nt) {
      v8f c0 = {}, c1 = {};
#pragma unroll
      for (int kt = 0; kt < 4; ++kt) {
        v16h b = load_b_frag(sW, lane, kt, nt);
        c0 = wmma_f16(a[0][kt], b, c0);
        c1 = wmma_f16(a[1][kt], b, c1);
      }
      const int n = (nt << 4) + cl;
      const float bb = sBmlp[n];
#pragma unroll
      for (int v = 0; v < 8; ++v) {
        const int eA = e0 + v + lh * 8;
        const int eB = eA + 16;
        if (eA < E) out[(size_t)eA * 128 + n] = c0[v] + bb;
        if (eB < E) out[(size_t)eB * 128 + n] = c1[v] + bb;
      }
    }
  }
}

extern "C" void kernel_launch(void* const* d_in, const int* in_sizes, int n_in,
                              void* d_out, int out_size, void* d_ws, size_t ws_size,
                              hipStream_t stream) {
  (void)n_in; (void)out_size; (void)d_ws; (void)ws_size;
  const float* emb   = (const float*)d_in[0];
  const int*   eidx  = (const int*)d_in[1];
  const float* Wsrc  = (const float*)d_in[2];
  const float* Wtgt  = (const float*)d_in[3];
  const float* Wproj = (const float*)d_in[4];
  const float* bproj = (const float*)d_in[5];
  const float* Wmlp  = (const float*)d_in[6];
  const float* bmlp  = (const float*)d_in[7];
  float* out = (float*)d_out;

  const int E = in_sizes[1] / 2;
  const int ntiles = (E + EDGES_PER_WAVE - 1) / EDGES_PER_WAVE;
  const int blocks = (ntiles + WAVES_PER_BLOCK - 1) / WAVES_PER_BLOCK;

  hipLaunchKernelGGL(MHAEdgeEmb_kernel, dim3(blocks), dim3(TPB), 0, stream,
                     emb, eidx, Wsrc, Wtgt, Wproj, bproj, Wmlp, bmlp, out, E, ntiles);
}
